// Head_38379827757258
// MI455X (gfx1250) — compile-verified, matrix operand order checked by reading, not run on hardware
//
#include <hip/hip_runtime.h>
#include <hip/hip_bf16.h>

#define BB 8
#define TT 2048
#define DD 1024
#define HH 128

typedef __attribute__((ext_vector_type(16))) __bf16 v16bf;
typedef __attribute__((ext_vector_type(8)))  float  v8f;

struct __align__(16) U4 { unsigned x, y, z, w; };
struct __align__(8)  S4 { unsigned short a, b, c, d; };

union Frag {
  v16bf v;
  U4    q[2];
};

// score scale (1024^-0.5) folded with log2(e): softmax done in base-2 domain
#define SC_LOG2E 0.045084440033810896f   // 0.03125 * 1.4426950408889634

__device__ __forceinline__ float fexp2(float x) {
  return __builtin_amdgcn_exp2f(x);      // v_exp_f32 (hardware base-2 exp)
}

__device__ __forceinline__ unsigned short f2bf(float f) {
  unsigned u = __float_as_uint(f);
  u += 0x7fffu + ((u >> 16) & 1u);   // round-to-nearest-even
  return (unsigned short)(u >> 16);
}

__device__ __forceinline__ U4 ldg16(const unsigned short* p) {
  return *(const U4*)p;
}

// ---------------------------------------------------------------------------
// Kernel 1: x (fp32) -> xb (bf16), 4 elements per thread
// ---------------------------------------------------------------------------
__global__ void cvt_x(const float* __restrict__ x,
                      unsigned short* __restrict__ xb, int n4) {
  int i = blockIdx.x * blockDim.x + threadIdx.x;
  if (i >= n4) return;
  const float4 f = ((const float4*)x)[i];
  S4 o;
  o.a = f2bf(f.x); o.b = f2bf(f.y); o.c = f2bf(f.z); o.d = f2bf(f.w);
  ((S4*)xb)[i] = o;
}

// ---------------------------------------------------------------------------
// Kernel 2: Wq/Wk/Wv [D][H] fp32 -> wt[p][h][d] bf16 (transposed)
// ---------------------------------------------------------------------------
__global__ void cvt_w(const float* __restrict__ Wq,
                      const float* __restrict__ Wk,
                      const float* __restrict__ Wv,
                      unsigned short* __restrict__ wt) {
  int i = blockIdx.x * blockDim.x + threadIdx.x;
  if (i >= 3 * HH * DD) return;
  int p = i / (HH * DD);
  int r = i % (HH * DD);
  int n = r / DD;       // head col
  int k = r % DD;       // embed dim
  const float* W = (p == 0) ? Wq : ((p == 1) ? Wk : Wv);
  wt[i] = f2bf(W[(size_t)k * HH + n]);
}

// ---------------------------------------------------------------------------
// Kernel 3: projections. One wave = one 16-row M tile of one projection.
// Q,K stored row-major bf16 [B*T][H]; V stored transposed bf16 [B][H][T].
// ---------------------------------------------------------------------------
__global__ void __launch_bounds__(128)
proj_qkv(const unsigned short* __restrict__ xb,
         const unsigned short* __restrict__ wt,
         unsigned short* __restrict__ qb,
         unsigned short* __restrict__ kb,
         unsigned short* __restrict__ vt) {
  const int lane  = threadIdx.x & 31;
  const int wid   = (blockIdx.x * blockDim.x + threadIdx.x) >> 5;
  const int projI = wid >> 10;          // 0..2
  const int mtile = wid & 1023;         // 0..1023
  const int row0  = mtile << 4;
  const int half  = lane >> 4;
  const int ln    = lane & 15;

  const unsigned short* xrow  = xb + (size_t)(row0 + ln) * DD;
  const unsigned short* wbase = wt + (size_t)projI * (HH * DD);

  v8f acc[8];
  const v8f vzero = {0.f,0.f,0.f,0.f,0.f,0.f,0.f,0.f};
#pragma unroll
  for (int i = 0; i < 8; ++i) acc[i] = vzero;

#pragma unroll 1
  for (int kc = 0; kc < DD / 32; ++kc) {
    const int kb0 = kc * 32 + half * 8;     // A-frag K pattern base
    Frag a;
    a.q[0] = ldg16(xrow + kb0);
    a.q[1] = ldg16(xrow + kb0 + 16);
#pragma unroll
    for (int nc = 0; nc < 8; ++nc) {
      const unsigned short* wrow = wbase + (size_t)(nc * 16 + ln) * DD;
      Frag bf;
      bf.q[0] = ldg16(wrow + kb0);
      bf.q[1] = ldg16(wrow + kb0 + 16);
      acc[nc] = __builtin_amdgcn_wmma_f32_16x16x32_bf16(
          false, a.v, false, bf.v, (short)0, acc[nc], false, false);
    }
  }

#pragma unroll
  for (int nc = 0; nc < 8; ++nc) {
#pragma unroll
    for (int i = 0; i < 8; ++i) {
      const int row = row0 + i + half * 8;
      const int col = nc * 16 + ln;
      const unsigned short val = f2bf(acc[nc][i]);
      if (projI == 0) {
        qb[(size_t)row * HH + col] = val;
      } else if (projI == 1) {
        kb[(size_t)row * HH + col] = val;
      } else {
        const int b = row >> 11, t = row & (TT - 1);
        vt[((size_t)b * HH + col) * TT + t] = val;
      }
    }
  }
}

// ---------------------------------------------------------------------------
// Kernel 4: causal flash attention. One wave = one 16-row q tile.
// 64-wide K blocks: 16 QK^T wmmas, ONE online-softmax pass (base-2 domain),
// P via per-wave LDS tile -> 16 PV wmmas.
// ---------------------------------------------------------------------------
__global__ void __launch_bounds__(128)
flash_attn(const unsigned short* __restrict__ qb,
           const unsigned short* __restrict__ kb,
           const unsigned short* __restrict__ vt,
           float* __restrict__ out) {
  __shared__ __align__(16) unsigned short Pldm[4][16 * 64];

  const int lane = threadIdx.x & 31;
  const int wIb  = threadIdx.x >> 5;
  const int wid  = (blockIdx.x * blockDim.x + threadIdx.x) >> 5;
  const int bI   = wid >> 7;            // batch
  const int qt   = wid & 127;           // q tile within batch
  const int r0   = qt << 4;
  const int half = lane >> 4;
  const int ln   = lane & 15;
  unsigned short* P = Pldm[wIb];

  // Q fragments (A layout): lane holds row r0+ln, 4 chunks of K=32 over H=128
  const unsigned short* Qr = qb + ((size_t)(bI * TT) + r0 + ln) * HH;
  Frag qf[4];
#pragma unroll
  for (int c = 0; c < 4; ++c) {
    const int kb0 = c * 32 + half * 8;
    qf[c].q[0] = ldg16(Qr + kb0);
    qf[c].q[1] = ldg16(Qr + kb0 + 16);
  }

  v8f o[8];
  float rm[8], rs[8];
  const v8f vzero = {0.f,0.f,0.f,0.f,0.f,0.f,0.f,0.f};
#pragma unroll
  for (int i = 0; i < 8; ++i) { o[i] = vzero; rm[i] = -1e30f; rs[i] = 0.f; }

  const int nblk = (r0 + 79) >> 6;      // 64-wide K blocks up to the diagonal
  for (int blk = 0; blk < nblk; ++blk) {
    const int kt0 = blk << 6;

    // ---- S = Q K^T (16x64, four 16-wide N tiles) ----
    v8f s[4];
#pragma unroll
    for (int nt = 0; nt < 4; ++nt) s[nt] = vzero;
#pragma unroll
    for (int nt = 0; nt < 4; ++nt) {
      const unsigned short* Kr =
          kb + ((size_t)(bI * TT) + kt0 + nt * 16 + ln) * HH;
#pragma unroll
      for (int c = 0; c < 4; ++c) {
        const int kb0 = c * 32 + half * 8;
        Frag kf;
        kf.q[0] = ldg16(Kr + kb0);
        kf.q[1] = ldg16(Kr + kb0 + 16);
        s[nt] = __builtin_amdgcn_wmma_f32_16x16x32_bf16(
            false, qf[c].v, false, kf.v, (short)0, s[nt], false, false);
      }
    }
    if (blk + 1 < nblk) {   // prefetch next K block -> global_prefetch_b8
      __builtin_prefetch(kb + ((size_t)(bI * TT) + kt0 + 64 + ln) * HH, 0, 1);
    }

    // ---- scale + causal mask + online softmax (base-2 domain) ----
#pragma unroll
    for (int i = 0; i < 8; ++i) {
      const int qrow = r0 + i + half * 8;
      float y[4];
#pragma unroll
      for (int j = 0; j < 4; ++j) {
        y[j] = (kt0 + 16 * j + ln <= qrow) ? s[j][i] * SC_LOG2E : -1e30f;
      }
      float t = fmaxf(fmaxf(y[0], y[1]), fmaxf(y[2], y[3]));
      t = fmaxf(t, __shfl_xor(t, 1, 32));
      t = fmaxf(t, __shfl_xor(t, 2, 32));
      t = fmaxf(t, __shfl_xor(t, 4, 32));
      t = fmaxf(t, __shfl_xor(t, 8, 32));
      const float nm   = fmaxf(rm[i], t);
      const float corr = fexp2(rm[i] - nm);
      float p[4], ps = 0.f;
#pragma unroll
      for (int j = 0; j < 4; ++j) {
        p[j] = fexp2(y[j] - nm);
        ps += p[j];
      }
      ps += __shfl_xor(ps, 1, 32);
      ps += __shfl_xor(ps, 2, 32);
      ps += __shfl_xor(ps, 4, 32);
      ps += __shfl_xor(ps, 8, 32);
      rs[i] = rs[i] * corr + ps;
      rm[i] = nm;
#pragma unroll
      for (int hc = 0; hc < 8; ++hc) o[hc][i] *= corr;
      // P (C layout) -> LDS row-major [16][64] bf16
#pragma unroll
      for (int j = 0; j < 4; ++j) {
        P[(i + half * 8) * 64 + 16 * j + ln] = f2bf(p[j]);
      }
    }
    asm volatile("s_wait_dscnt 0" ::: "memory");

    // reload P as A fragments (lane = row; two K=32 chunks over 64 cols)
    Frag pa0, pa1;
    pa0.q[0] = *(const U4*)&P[ln * 64 + half * 8];
    pa0.q[1] = *(const U4*)&P[ln * 64 + half * 8 + 16];
    pa1.q[0] = *(const U4*)&P[ln * 64 + 32 + half * 8];
    pa1.q[1] = *(const U4*)&P[ln * 64 + 32 + half * 8 + 16];

    // ---- O += P * V (V transposed: rows of Vt contiguous along t) ----
#pragma unroll
    for (int hc = 0; hc < 8; ++hc) {
      const unsigned short* Vr =
          vt + ((size_t)(bI * HH) + hc * 16 + ln) * TT + kt0;
      Frag vf0, vf1;
      vf0.q[0] = ldg16(Vr + half * 8);
      vf0.q[1] = ldg16(Vr + half * 8 + 16);
      vf1.q[0] = ldg16(Vr + 32 + half * 8);
      vf1.q[1] = ldg16(Vr + 32 + half * 8 + 16);
      o[hc] = __builtin_amdgcn_wmma_f32_16x16x32_bf16(
          false, pa0.v, false, vf0.v, (short)0, o[hc], false, false);
      o[hc] = __builtin_amdgcn_wmma_f32_16x16x32_bf16(
          false, pa1.v, false, vf1.v, (short)0, o[hc], false, false);
    }
  }

  // ---- epilogue: divide by row sums, store fp32 ----
#pragma unroll
  for (int i = 0; i < 8; ++i) {
    const float inv = 1.0f / rs[i];
    const int row   = r0 + i + half * 8;
#pragma unroll
    for (int hc = 0; hc < 8; ++hc) {
      out[((size_t)(bI * TT) + row) * HH + hc * 16 + ln] = o[hc][i] * inv;
    }
  }
}

// ---------------------------------------------------------------------------
extern "C" void kernel_launch(void* const* d_in, const int* in_sizes, int n_in,
                              void* d_out, int out_size, void* d_ws, size_t ws_size,
                              hipStream_t stream) {
  const float* x  = (const float*)d_in[0];
  const float* Wq = (const float*)d_in[1];
  const float* Wk = (const float*)d_in[2];
  const float* Wv = (const float*)d_in[3];
  float* out = (float*)d_out;

  unsigned short* xb = (unsigned short*)d_ws;               // [B*T][D] bf16
  unsigned short* wt = xb + (size_t)BB * TT * DD;           // [3][H][D] bf16
  unsigned short* qb = wt + (size_t)3 * HH * DD;            // [B*T][H] bf16
  unsigned short* kb = qb + (size_t)BB * TT * HH;           // [B*T][H] bf16
  unsigned short* vt = kb + (size_t)BB * TT * HH;           // [B][H][T] bf16

  {
    int n4 = BB * TT * DD / 4;
    cvt_x<<<(n4 + 255) / 256, 256, 0, stream>>>(x, xb, n4);
  }
  {
    int n = 3 * HH * DD;
    cvt_w<<<(n + 255) / 256, 256, 0, stream>>>(Wq, Wk, Wv, wt);
  }
  // 3 projections * 1024 M-tiles = 3072 waves, 4 waves/block
  proj_qkv<<<768, 128, 0, stream>>>(xb, wt, qb, kb, vt);
  // 8 batches * 128 q-tiles = 1024 waves, 4 waves/block
  flash_attn<<<256, 128, 0, stream>>>(qb, kb, vt, out);
}
